// GCN2_35974646071761
// MI455X (gfx1250) — matrix operand order repaired and checked by reading, not compiled, not run on hardware
//
#include <hip/hip_runtime.h>
#include <hip/hip_bf16.h>

#define NN 50000
#define NE 800000
#define NF 256
#define NH 128
#define NC 40
#define NCP 48   // NC padded to WMMA tile multiple

typedef __attribute__((ext_vector_type(16))) _Float16 v16h;
typedef __attribute__((ext_vector_type(8)))  float    v8f;

// ---------------------------------------------------------------------------
// Zero the accumulation buffers (emb region of d_out, z accumulator in ws).
// ---------------------------------------------------------------------------
__global__ __launch_bounds__(256) void gcn_zero_kernel(float* __restrict__ emb,
                                                       float* __restrict__ zacc) {
    long i = (long)blockIdx.x * 256 + threadIdx.x;
    if (i < (long)NN * NH) emb[i] = 0.0f;
    if (i < (long)NN * NC) zacc[i] = 0.0f;
}

// ---------------------------------------------------------------------------
// Build an A fragment (16x32 f16, contiguous k order, matched by B) from a row-major f32 row:
// this lane holds K = kb..kb+15 (kb = kt*32 + 16*hi).
// ---------------------------------------------------------------------------
__device__ __forceinline__ v16h load_a_frag(const float* __restrict__ xr, int kb) {
    const float4 f0 = *(const float4*)(xr + kb);
    const float4 f1 = *(const float4*)(xr + kb + 4);
    const float4 f2 = *(const float4*)(xr + kb + 8);
    const float4 f3 = *(const float4*)(xr + kb + 12);
    v16h a;
    a[0]  = (_Float16)f0.x; a[1]  = (_Float16)f0.y; a[2]  = (_Float16)f0.z; a[3]  = (_Float16)f0.w;
    a[4]  = (_Float16)f1.x; a[5]  = (_Float16)f1.y; a[6]  = (_Float16)f1.z; a[7]  = (_Float16)f1.w;
    a[8]  = (_Float16)f2.x; a[9]  = (_Float16)f2.y; a[10] = (_Float16)f2.z; a[11] = (_Float16)f2.w;
    a[12] = (_Float16)f3.x; a[13] = (_Float16)f3.y; a[14] = (_Float16)f3.z; a[15] = (_Float16)f3.w;
    return a;
}

// ---------------------------------------------------------------------------
// GEMM1: h0[NN,NH] = x[NN,NF] @ W1[NF,NH], f16 WMMA with f32 accumulate.
// Block = 256 threads = 8 wave32; each wave computes a 16x128 output strip.
// W1 staged in LDS transposed (col-major, f16). Per k-tile, all 8 B fragments
// are loaded into a register array first (one DS clause, staggered waits),
// then 8 WMMAs issue back-to-back. __launch_bounds__(256,1) releases the
// register budget so the b[] array stays live (software pipeline intact).
// ---------------------------------------------------------------------------
__global__ __launch_bounds__(256, 1) void gcn_gemm1_kernel(const float* __restrict__ x,
                                                           const float* __restrict__ W1,
                                                           float* __restrict__ h0) {
    __shared__ _Float16 bt[NH * NF];            // 64 KB: bt[n*NF + k]
    for (int idx = threadIdx.x; idx < NH * NF; idx += 256) {
        int n = idx & (NH - 1);
        int k = idx >> 7;
        bt[n * NF + k] = (_Float16)W1[k * NH + n];
    }
    __syncthreads();

    const int lane = threadIdx.x & 31;
    const int wid  = threadIdx.x >> 5;
    const int m0   = blockIdx.x * 128 + wid * 16;
    const int hi   = (lane & 16) ? 1 : 0;       // which half-wave

    int row = m0 + (lane & 15);
    if (row >= NN) row = NN - 1;                // clamp: EXEC must stay all-ones
    const float* xr = x + (long)row * NF;

    // Per-lane LDS base for B fragments: column (lane&15), K offset 16*hi.
    const _Float16* bbase = &bt[(lane & 15) * NF + hi * 16];

    v8f acc[8] = {};

#pragma unroll
    for (int kt = 0; kt < 8; ++kt) {
        const v16h a = load_a_frag(xr, kt * 32 + hi * 16);

        v16h b[8];
#pragma unroll
        for (int nt = 0; nt < 8; ++nt)
            b[nt] = *(const v16h*)(bbase + nt * 16 * NF + kt * 32);

#pragma unroll
        for (int nt = 0; nt < 8; ++nt)
            acc[nt] = __builtin_amdgcn_wmma_f32_16x16x32_f16(
                false, a, false, b[nt], (short)0, acc[nt], false, false);
    }

    // C/D layout: VGPR r -> row m0 + r + 8*hi, lane&15 -> column
    const int rb = m0 + hi * 8;
#pragma unroll
    for (int nt = 0; nt < 8; ++nt) {
        const int col = nt * 16 + (lane & 15);
#pragma unroll
        for (int r = 0; r < 8; ++r) {
            const int rr = rb + r;
            if (rr < NN) h0[(long)rr * NH + col] = acc[nt][r];
        }
    }
}

// ---------------------------------------------------------------------------
// SPMM1: emb[d] += w_e * h0[s] for each edge, feature dim 128.
// One wave per edge, 4 floats per lane, f32 global atomics (L2-resident).
// ---------------------------------------------------------------------------
__global__ __launch_bounds__(256) void gcn_spmm1_kernel(const int* __restrict__ src,
                                                        const int* __restrict__ dst,
                                                        const float* __restrict__ w,
                                                        const float* __restrict__ h0,
                                                        float* __restrict__ emb) {
    const int e = blockIdx.x * 8 + (threadIdx.x >> 5);
    if (e >= NE) return;
    const int l4 = (threadIdx.x & 31) * 4;
    const int s = src[e];
    const int d = dst[e];
    const float we = w[e];
    const float4 v = *(const float4*)(h0 + (long)s * NH + l4);
    float* o = emb + (long)d * NH + l4;
    atomicAdd(o + 0, v.x * we);
    atomicAdd(o + 1, v.y * we);
    atomicAdd(o + 2, v.z * we);
    atomicAdd(o + 3, v.w * we);
}

// ---------------------------------------------------------------------------
// emb = relu(emb + b1), in place.
// ---------------------------------------------------------------------------
__global__ __launch_bounds__(256) void gcn_bias_relu_kernel(float* __restrict__ emb,
                                                            const float* __restrict__ b1) {
    long i = (long)blockIdx.x * 256 + threadIdx.x;
    if (i >= (long)NN * NH) return;
    float v = emb[i] + b1[i & (NH - 1)];
    emb[i] = v > 0.0f ? v : 0.0f;
}

// ---------------------------------------------------------------------------
// GEMM2: z0[NN,NC] = emb[NN,NH] @ W2[NH,NC], N padded 40->48 with zero cols.
// All 12 B fragments (entire W2) are hoisted into registers before the
// k-loop: the main loop has zero LDS traffic.
// ---------------------------------------------------------------------------
__global__ __launch_bounds__(256, 1) void gcn_gemm2_kernel(const float* __restrict__ emb,
                                                           const float* __restrict__ W2,
                                                           float* __restrict__ z0) {
    __shared__ _Float16 bt[NCP * NH];           // 12 KB: bt[n*NH + k]
    for (int idx = threadIdx.x; idx < NCP * NH; idx += 256) {
        int n = idx >> 7;
        int k = idx & (NH - 1);
        float v = (n < NC) ? W2[k * NC + n] : 0.0f;
        bt[n * NH + k] = (_Float16)v;
    }
    __syncthreads();

    const int lane = threadIdx.x & 31;
    const int wid  = threadIdx.x >> 5;
    const int m0   = blockIdx.x * 128 + wid * 16;
    const int hi   = (lane & 16) ? 1 : 0;

    int row = m0 + (lane & 15);
    if (row >= NN) row = NN - 1;
    const float* xr = emb + (long)row * NH;

    // Hoist all of W2 into registers: b[kt][nt], 12 fragments = 96 VGPRs.
    const _Float16* bbase = &bt[(lane & 15) * NH + hi * 16];
    v16h b[4][3];
#pragma unroll
    for (int kt = 0; kt < 4; ++kt)
#pragma unroll
        for (int nt = 0; nt < 3; ++nt)
            b[kt][nt] = *(const v16h*)(bbase + nt * 16 * NH + kt * 32);

    v8f acc[3] = {};

#pragma unroll
    for (int kt = 0; kt < 4; ++kt) {
        const v16h a = load_a_frag(xr, kt * 32 + hi * 16);
#pragma unroll
        for (int nt = 0; nt < 3; ++nt)
            acc[nt] = __builtin_amdgcn_wmma_f32_16x16x32_f16(
                false, a, false, b[kt][nt], (short)0, acc[nt], false, false);
    }

    const int rb = m0 + hi * 8;
#pragma unroll
    for (int nt = 0; nt < 3; ++nt) {
        const int col = nt * 16 + (lane & 15);
        if (col >= NC) continue;
#pragma unroll
        for (int r = 0; r < 8; ++r) {
            const int rr = rb + r;
            if (rr < NN) z0[(long)rr * NC + col] = acc[nt][r];
        }
    }
}

// ---------------------------------------------------------------------------
// SPMM2: zacc[d] += w_e * z0[s], feature dim 40. Thread per (edge, class).
// ---------------------------------------------------------------------------
__global__ __launch_bounds__(256) void gcn_spmm2_kernel(const int* __restrict__ src,
                                                        const int* __restrict__ dst,
                                                        const float* __restrict__ w,
                                                        const float* __restrict__ z0,
                                                        float* __restrict__ zacc) {
    long t = (long)blockIdx.x * 256 + threadIdx.x;
    if (t >= (long)NE * NC) return;
    int e = (int)(t / NC);
    int c = (int)(t - (long)e * NC);
    float v = z0[(long)src[e] * NC + c] * w[e];
    atomicAdd(&zacc[(long)dst[e] * NC + c], v);
}

// ---------------------------------------------------------------------------
// out = log_softmax(zacc + b2) over 40 classes, one thread per node.
// ---------------------------------------------------------------------------
__global__ __launch_bounds__(256) void gcn_logsoftmax_kernel(const float* __restrict__ zacc,
                                                             const float* __restrict__ b2,
                                                             float* __restrict__ out) {
    int n = blockIdx.x * 256 + threadIdx.x;
    if (n >= NN) return;
    float v[NC];
    float m = -3.402823466e38f;
#pragma unroll
    for (int c = 0; c < NC; ++c) {
        v[c] = zacc[(long)n * NC + c] + b2[c];
        m = fmaxf(m, v[c]);
    }
    float s = 0.0f;
#pragma unroll
    for (int c = 0; c < NC; ++c) s += __expf(v[c] - m);
    float ls = __logf(s);
#pragma unroll
    for (int c = 0; c < NC; ++c) out[(long)n * NC + c] = v[c] - m - ls;
}

// ---------------------------------------------------------------------------
// Launch: zero -> gemm1 -> spmm1 -> bias_relu -> gemm2 -> spmm2 -> logsoftmax
// d_out layout: [logits NN*NC][emb NN*NH]. ws: [h0 / z0 reuse][zacc].
// ---------------------------------------------------------------------------
extern "C" void kernel_launch(void* const* d_in, const int* in_sizes, int n_in,
                              void* d_out, int out_size, void* d_ws, size_t ws_size,
                              hipStream_t stream) {
    const float* x    = (const float*)d_in[0];
    const int*   esrc = (const int*)d_in[1];
    const int*   edst = (const int*)d_in[2];
    const float* ew   = (const float*)d_in[3];
    const float* W1   = (const float*)d_in[4];
    const float* b1   = (const float*)d_in[5];
    const float* W2   = (const float*)d_in[6];
    const float* b2   = (const float*)d_in[7];

    float* logits = (float*)d_out;                  // [NN, NC]
    float* emb    = (float*)d_out + (long)NN * NC;  // [NN, NH]

    float* ws   = (float*)d_ws;
    float* h0   = ws;                               // [NN, NH]
    float* z0   = ws;                               // [NN, NC] (reuses dead h0)
    float* zacc = ws + (long)NN * NH;               // [NN, NC]

    const int gemm_blocks = (NN + 127) / 128;       // 391

    gcn_zero_kernel<<<(int)(((long)NN * NH + 255) / 256), 256, 0, stream>>>(emb, zacc);
    gcn_gemm1_kernel<<<gemm_blocks, 256, 0, stream>>>(x, W1, h0);
    gcn_spmm1_kernel<<<(NE + 7) / 8, 256, 0, stream>>>(esrc, edst, ew, h0, emb);
    gcn_bias_relu_kernel<<<(int)(((long)NN * NH + 255) / 256), 256, 0, stream>>>(emb, b1);
    gcn_gemm2_kernel<<<gemm_blocks, 256, 0, stream>>>(emb, W2, z0);
    gcn_spmm2_kernel<<<(int)(((long)NE * NC + 255) / 256), 256, 0, stream>>>(esrc, edst, ew, z0, zacc);
    gcn_logsoftmax_kernel<<<(NN + 255) / 256, 256, 0, stream>>>(zacc, b2, logits);
}
